// MyModel_7971459302017
// MI455X (gfx1250) — compile-verified, matrix operand order checked by reading, not run on hardware
//
#include <hip/hip_runtime.h>

typedef _Float16 h16;
typedef __attribute__((ext_vector_type(16))) _Float16 v16h;
typedef __attribute__((ext_vector_type(8)))  _Float16 v8h;
typedef __attribute__((ext_vector_type(8)))  float    v8f;
typedef int v4i_vs __attribute__((vector_size(16)));   // matches builtin param

// ---- optional CDNA5 async global->LDS path (ASYNCcnt), compile-safe probe ----
#if __has_builtin(__builtin_amdgcn_global_load_async_to_lds_b128) && \
    __has_builtin(__builtin_amdgcn_s_wait_asynccnt)
#define HAVE_ASYNC_LDS 1
#endif

static __device__ __forceinline__ void copy_g2l_b128(h16* lds_dst, const h16* gsrc) {
#ifdef HAVE_ASYNC_LDS
  __builtin_amdgcn_global_load_async_to_lds_b128(
      (v4i_vs*)gsrc, (v4i_vs*)lds_dst, 0, 0);
#else
  *(v8h*)lds_dst = *(const v8h*)gsrc;
#endif
}
static __device__ __forceinline__ void lds_zero_b128(h16* lds_dst) {
  v8h z = {};
  *(v8h*)lds_dst = z;
}
static __device__ __forceinline__ void tile_sync() {
#ifdef HAVE_ASYNC_LDS
  __builtin_amdgcn_s_wait_asynccnt(0);
#endif
  __syncthreads();
}

// Build a 16-element A/B fragment for v_wmma_f32_16x16x32_f16 from an LDS row
// of 32 contiguous K-halfs. Per ISA 7.12.2 (16-bit A 16x32): lanes 0-15 hold
// K[0..7] (elems 0-7) and K[16..23] (elems 8-15); lanes 16-31 hold K[8..15]
// and K[24..31].  kh = (lane>=16)?8:0.  Two ds_load_b128 per fragment.
static __device__ __forceinline__ v16h frag16(const h16* rowbase, int kh) {
  v8h lo = *(const v8h*)(rowbase + kh);
  v8h hi = *(const v8h*)(rowbase + kh + 16);
  return __builtin_shufflevector(lo, hi, 0,1,2,3,4,5,6,7,8,9,10,11,12,13,14,15);
}
#define WMMA_F16(A, B, C) \
  __builtin_amdgcn_wmma_f32_16x16x32_f16(false, (A), false, (B), (short)0, (C), false, false)

// ---------------------------------------------------------------------------
// 3x3 SAME conv as 9 shifted 1x1 GEMMs.  in: NHWC f16 (Cin mult of 32),
// wt: f16 [tap][Cout][Cin] (B pre-transposed), out: NHWC f16.
// Block 256 = 8 waves; block tile 128(M) x 64(N); wave tile 32x32 -> 4 WMMA
// per K-chunk per wave.
// ---------------------------------------------------------------------------
template<bool RELU>
__global__ __launch_bounds__(256)
void conv3x3_wmma(const h16* __restrict__ in, const h16* __restrict__ wt,
                  const float* __restrict__ bias, h16* __restrict__ out,
                  int Bn, int H, int W, int Cin, int Cout)
{
  __shared__ __align__(16) h16 As[128 * 32];
  __shared__ __align__(16) h16 Bs[64 * 32];
  const int tid  = threadIdx.x;
  const int lane = tid & 31;
  const int wv   = tid >> 5;
  const int tm   = wv >> 1;      // 0..3  (32-row band)
  const int tn   = wv & 1;       // 0..1  (32-col band)
  const int mtile = blockIdx.x, ntile = blockIdx.y;
  const long npix = (long)Bn * H * W;
  const int kch = Cin >> 5;

  v8f acc00 = {}, acc01 = {}, acc10 = {}, acc11 = {};
  for (int tap = 0; tap < 9; ++tap) {
    const int dy = tap / 3 - 1, dx = tap % 3 - 1;
    for (int kc = 0; kc < kch; ++kc) {
      { // A tile: 128 rows x 32 halfs; thread -> one row, 16 halfs (2 x b128)
        int row = tid >> 1, seg0 = (tid & 1) << 1;
        long gp = (long)mtile * 128 + row;
        const h16* src = nullptr;
        if (gp < npix) {
          int w = (int)(gp % W); long t = gp / W;
          int h = (int)(t % H); int b = (int)(t / H);
          int hs = h + dy, ws = w + dx;
          if ((unsigned)hs < (unsigned)H && (unsigned)ws < (unsigned)W)
            src = in + ((size_t)(b * H + hs) * W + ws) * Cin
                     + (kc << 5) + (seg0 << 3);
        }
        h16* dst = As + row * 32 + (seg0 << 3);
        if (src) { copy_g2l_b128(dst, src); copy_g2l_b128(dst + 8, src + 8); }
        else     { lds_zero_b128(dst);      lds_zero_b128(dst + 8); }
      }
      { // B tile: 64 n-rows x 32 halfs (bT layout [n][k]); thread -> one b128
        int n = tid >> 2, seg = tid & 3;
        int oc = ntile * 64 + n;
        h16* dst = Bs + n * 32 + (seg << 3);
        if (oc < Cout)
          copy_g2l_b128(dst, wt + ((size_t)tap * Cout + oc) * Cin
                                + (kc << 5) + (seg << 3));
        else
          lds_zero_b128(dst);
      }
      tile_sync();
      const int kh = (lane >> 4) << 3;
      v16h a0 = frag16(As + (tm * 32 +      (lane & 15)) * 32, kh);
      v16h a1 = frag16(As + (tm * 32 + 16 + (lane & 15)) * 32, kh);
      v16h b0 = frag16(Bs + (tn * 32 +      (lane & 15)) * 32, kh);
      v16h b1 = frag16(Bs + (tn * 32 + 16 + (lane & 15)) * 32, kh);
      acc00 = WMMA_F16(a0, b0, acc00);
      acc01 = WMMA_F16(a0, b1, acc01);
      acc10 = WMMA_F16(a1, b0, acc10);
      acc11 = WMMA_F16(a1, b1, acc11);
      __syncthreads();
    }
  }
  // C/D layout: VGPR v, lanes 0-15 -> M=v, lanes 16-31 -> M=8+v; N = lane&15.
  const int  colb = ntile * 64 + tn * 32 + (lane & 15);
  const long rowb = (long)mtile * 128 + tm * 32 + ((lane >> 4) << 3);
  auto store_tile = [&](const v8f& a, long r0, int c0) {
    if (c0 < Cout) {
      const float bv = bias[c0];
      for (int v = 0; v < 8; ++v) {
        long gp = r0 + v;
        if (gp < npix) {
          float val = a[v] + bv;
          if (RELU) val = fmaxf(val, 0.f);
          out[(size_t)gp * Cout + c0] = (h16)val;
        }
      }
    }
  };
  store_tile(acc00, rowb,      colb);
  store_tile(acc01, rowb,      colb + 16);
  store_tile(acc10, rowb + 16, colb);
  store_tile(acc11, rowb + 16, colb + 16);
}

// ---------------------------------------------------------------------------
// Plain GEMM: out[M,N] f32 = A[M,K] f16 * Bw[N,K]^T f16 (+bias)
// ---------------------------------------------------------------------------
__global__ __launch_bounds__(256)
void gemm_wmma(const h16* __restrict__ A, const h16* __restrict__ Bw,
               const float* __restrict__ bias, float* __restrict__ out,
               int M, int N, int K)
{
  __shared__ __align__(16) h16 As[128 * 32];
  __shared__ __align__(16) h16 Bs[64 * 32];
  const int tid = threadIdx.x, lane = tid & 31, wv = tid >> 5;
  const int tm = wv >> 1, tn = wv & 1;
  const int mtile = blockIdx.x, ntile = blockIdx.y;
  const int kch = K >> 5;
  v8f acc00 = {}, acc01 = {}, acc10 = {}, acc11 = {};
  for (int kc = 0; kc < kch; ++kc) {
    {
      int row = tid >> 1, seg0 = (tid & 1) << 1;
      long gp = (long)mtile * 128 + row;
      h16* dst = As + row * 32 + (seg0 << 3);
      if (gp < M) {
        const h16* src = A + (size_t)gp * K + (kc << 5) + (seg0 << 3);
        __builtin_prefetch(src + 32, 0, 1);
        copy_g2l_b128(dst, src); copy_g2l_b128(dst + 8, src + 8);
      } else {
        lds_zero_b128(dst); lds_zero_b128(dst + 8);
      }
    }
    {
      int nn = tid >> 2, seg = tid & 3;
      int oc = ntile * 64 + nn;
      h16* dst = Bs + nn * 32 + (seg << 3);
      if (oc < N)
        copy_g2l_b128(dst, Bw + (size_t)oc * K + (kc << 5) + (seg << 3));
      else
        lds_zero_b128(dst);
    }
    tile_sync();
    const int kh = (lane >> 4) << 3;
    v16h a0 = frag16(As + (tm * 32 +      (lane & 15)) * 32, kh);
    v16h a1 = frag16(As + (tm * 32 + 16 + (lane & 15)) * 32, kh);
    v16h b0 = frag16(Bs + (tn * 32 +      (lane & 15)) * 32, kh);
    v16h b1 = frag16(Bs + (tn * 32 + 16 + (lane & 15)) * 32, kh);
    acc00 = WMMA_F16(a0, b0, acc00);
    acc01 = WMMA_F16(a0, b1, acc01);
    acc10 = WMMA_F16(a1, b0, acc10);
    acc11 = WMMA_F16(a1, b1, acc11);
    __syncthreads();
  }
  const int  colb = ntile * 64 + tn * 32 + (lane & 15);
  const long rowb = (long)mtile * 128 + tm * 32 + ((lane >> 4) << 3);
  auto store_tile = [&](const v8f& a, long r0, int c0) {
    if (c0 < N) {
      const float bv = bias ? bias[c0] : 0.f;
      for (int v = 0; v < 8; ++v) {
        long gp = r0 + v;
        if (gp < M) out[(size_t)gp * N + c0] = a[v] + bv;
      }
    }
  };
  store_tile(acc00, rowb,      colb);
  store_tile(acc01, rowb,      colb + 16);
  store_tile(acc10, rowb + 16, colb);
  store_tile(acc11, rowb + 16, colb + 16);
}

// ---------------------------------------------------------------------------
// ConvTranspose 2x2 stride 2: out[b,2h+p,2w+q,o] = in[b,h,w,:] . wt[tap][o][:]
// blockIdx.z = tap (p*2+q).  in: NHWC f16 (H,W input dims), out: (2H,2W,Cout).
// ---------------------------------------------------------------------------
__global__ __launch_bounds__(256)
void convt2x2_wmma(const h16* __restrict__ in, const h16* __restrict__ wt,
                   const float* __restrict__ bias, h16* __restrict__ out,
                   int Bn, int H, int W, int Cin, int Cout)
{
  __shared__ __align__(16) h16 As[128 * 32];
  __shared__ __align__(16) h16 Bs[64 * 32];
  const int tid = threadIdx.x, lane = tid & 31, wv = tid >> 5;
  const int tm = wv >> 1, tn = wv & 1;
  const int mtile = blockIdx.x, ntile = blockIdx.y, tap = blockIdx.z;
  const int p = tap >> 1, q = tap & 1;
  const long npix = (long)Bn * H * W;
  const int kch = Cin >> 5;
  v8f acc00 = {}, acc01 = {}, acc10 = {}, acc11 = {};
  for (int kc = 0; kc < kch; ++kc) {
    {
      int row = tid >> 1, seg0 = (tid & 1) << 1;
      long gp = (long)mtile * 128 + row;
      h16* dst = As + row * 32 + (seg0 << 3);
      if (gp < npix) {
        const h16* src = in + (size_t)gp * Cin + (kc << 5) + (seg0 << 3);
        copy_g2l_b128(dst, src); copy_g2l_b128(dst + 8, src + 8);
      } else {
        lds_zero_b128(dst); lds_zero_b128(dst + 8);
      }
    }
    {
      int nn = tid >> 2, seg = tid & 3;
      int oc = ntile * 64 + nn;
      h16* dst = Bs + nn * 32 + (seg << 3);
      if (oc < Cout)
        copy_g2l_b128(dst, wt + ((size_t)tap * Cout + oc) * Cin
                              + (kc << 5) + (seg << 3));
      else
        lds_zero_b128(dst);
    }
    tile_sync();
    const int kh = (lane >> 4) << 3;
    v16h a0 = frag16(As + (tm * 32 +      (lane & 15)) * 32, kh);
    v16h a1 = frag16(As + (tm * 32 + 16 + (lane & 15)) * 32, kh);
    v16h b0 = frag16(Bs + (tn * 32 +      (lane & 15)) * 32, kh);
    v16h b1 = frag16(Bs + (tn * 32 + 16 + (lane & 15)) * 32, kh);
    acc00 = WMMA_F16(a0, b0, acc00);
    acc01 = WMMA_F16(a0, b1, acc01);
    acc10 = WMMA_F16(a1, b0, acc10);
    acc11 = WMMA_F16(a1, b1, acc11);
    __syncthreads();
  }
  const int  colb = ntile * 64 + tn * 32 + (lane & 15);
  const long rowb = (long)mtile * 128 + tm * 32 + ((lane >> 4) << 3);
  auto store_tile = [&](const v8f& a, long r0, int c0) {
    if (c0 < Cout) {
      const float bv = bias[c0];
      for (int v = 0; v < 8; ++v) {
        long gp = r0 + v;
        if (gp < npix) {
          int w = (int)(gp % W); long t = gp / W;
          int h = (int)(t % H); int b = (int)(t / H);
          size_t oidx = (((size_t)b * 2 * H + 2 * h + p) * (2 * W) + 2 * w + q)
                          * Cout + c0;
          out[oidx] = (h16)(a[v] + bv);
        }
      }
    }
  };
  store_tile(acc00, rowb,      colb);
  store_tile(acc01, rowb,      colb + 16);
  store_tile(acc10, rowb + 16, colb);
  store_tile(acc11, rowb + 16, colb + 16);
}

// --------------------------- scalar helper kernels -------------------------

// x (B,5,16,192) NCHW f32 -> NHWC f16, channels padded 5->32 with zeros
__global__ void pack_x(const float* __restrict__ x, h16* __restrict__ xp, int total) {
  int i = blockIdx.x * 256 + threadIdx.x; if (i >= total) return;
  int c = i & 31; int pix = i >> 5;
  int w = pix % 192; int t = pix / 192; int h = t % 16; int b = t / 16;
  float v = 0.f;
  if (c < 5) v = x[(((size_t)b * 5 + c) * 16 + h) * 192 + w];
  xp[i] = (h16)v;
}

// OIHW f32 (O,I,3,3) -> f16 [tap][O][Ipad], zero-padded I -> Ipad
__global__ void pack_conv3(const float* __restrict__ w, h16* __restrict__ o,
                           int O, int I, int Ip, int total) {
  int idx = blockIdx.x * 256 + threadIdx.x; if (idx >= total) return;
  int i = idx % Ip; int r = idx / Ip; int oc = r % O; int t = r / O;
  float v = (i < I) ? w[((size_t)oc * I + i) * 9 + t] : 0.f;
  o[idx] = (h16)v;
}

// copq f32 (C,O,2,2) -> f16 [tap][O][C]
__global__ void pack_convt(const float* __restrict__ w, h16* __restrict__ o,
                           int C, int O, int total) {
  int idx = blockIdx.x * 256 + threadIdx.x; if (idx >= total) return;
  int c = idx % C; int r = idx / C; int oc = r % O; int t = r / O;
  int p = t >> 1, q = t & 1;
  o[idx] = (h16)w[(((size_t)c * O + oc) * 2 + p) * 2 + q];
}

// graph W (64, 2C) f32 -> f16 [128][C]: rows 0..63 = W1-W2, rows 64..127 = W2
__global__ void pack_qr(const float* __restrict__ g, h16* __restrict__ o,
                        int C, int total) {
  int idx = blockIdx.x * 256 + threadIdx.x; if (idx >= total) return;
  int c = idx % C; int n = idx / C;
  float v = (n < 64) ? (g[(size_t)n * 2 * C + c] - g[(size_t)n * 2 * C + C + c])
                     : g[(size_t)(n - 64) * 2 * C + C + c];
  o[idx] = (h16)v;
}

__global__ void cvt16(const float* __restrict__ a, h16* __restrict__ o, int total) {
  int i = blockIdx.x * 256 + threadIdx.x; if (i >= total) return;
  o[i] = (h16)a[i];
}

// 2x2 maxpool NHWC f16 -> NHWC f16 (+ optional f32 NCHW copy)
__global__ void maxpool2h(const h16* __restrict__ in, h16* __restrict__ out,
                          float* __restrict__ out32,
                          int Bn, int H, int W, int C, int total) {
  int i = blockIdx.x * 256 + threadIdx.x; if (i >= total) return;
  int c = i % C; int r = i / C;
  int W2 = W >> 1, H2 = H >> 1;
  int w2 = r % W2; r /= W2; int h2 = r % H2; int b = r / H2;
  const h16* p0 = in + (((size_t)b * H + 2 * h2) * W + 2 * w2) * C + c;
  float m = (float)p0[0];
  m = fmaxf(m, (float)p0[C]);
  m = fmaxf(m, (float)p0[(size_t)W * C]);
  m = fmaxf(m, (float)p0[(size_t)W * C + C]);
  out[i] = (h16)m;
  if (out32) out32[(((size_t)b * C + c) * H2 + h2) * W2 + w2] = m;
}

__global__ void sqnorm(const h16* __restrict__ f, float* __restrict__ sq,
                       int total, int C) {
  int i = blockIdx.x * 256 + threadIdx.x; if (i >= total) return;
  const h16* p = f + (size_t)i * C;
  float s = 0.f;
  for (int c = 0; c < C; ++c) { float v = (float)p[c]; s += v * v; }
  sq[i] = s;
}

// kNN(9) + edge-conv using q/r decomposition:
// out[n,o] = relu( q[n,o] + max_k r[knn(n)_k, o] + b[o] )
__global__ __launch_bounds__(64)
void graph_edge(const h16* __restrict__ feat, const float* __restrict__ sq,
                const float* __restrict__ qr, const float* __restrict__ gb,
                float* __restrict__ gout, int Np, int C)
{
  __shared__ __align__(16) h16 myf[64 * 128];
  const int b = blockIdx.y;
  const int n = blockIdx.x * 64 + threadIdx.x;
  const h16* fb = feat + (size_t)b * Np * C;
  h16* mine = myf + threadIdx.x * C;
  for (int c = 0; c < C; c += 8)
    *(v8h*)(mine + c) = *(const v8h*)(fb + (size_t)n * C + c);

  float best[9]; int bidx[9];
  #pragma unroll
  for (int j = 0; j < 9; ++j) { best[j] = 3.0e38f; bidx[j] = 0; }
  const float sqn = sq[(size_t)b * Np + n];
  for (int m = 0; m < Np; ++m) {
    const h16* fm = fb + (size_t)m * C;
    float dot = 0.f;
    for (int c = 0; c < C; c += 8) {
      v8h a  = *(const v8h*)(mine + c);
      v8h bb = *(const v8h*)(fm + c);
      #pragma unroll
      for (int e = 0; e < 8; ++e) dot += (float)a[e] * (float)bb[e];
    }
    float d = sqn + sq[(size_t)b * Np + m] - 2.f * dot;
    if (d < best[8]) {                     // stable insertion (ties keep low idx)
      int j = 8;
      while (j > 0 && d < best[j - 1]) {
        best[j] = best[j - 1]; bidx[j] = bidx[j - 1]; --j;
      }
      best[j] = d; bidx[j] = m;
    }
  }
  const float* qrow = qr + ((size_t)b * Np + n) * 128;
  for (int o = 0; o < 64; ++o) {
    float mx = -3.0e38f;
    #pragma unroll
    for (int k = 0; k < 9; ++k)
      mx = fmaxf(mx, qr[((size_t)b * Np + bidx[k]) * 128 + 64 + o]);
    float val = qrow[o] + mx + gb[o];
    gout[((size_t)b * 64 + o) * Np + n] = fmaxf(val, 0.f);
  }
}

// x1g (B,64,8,96) f32 -> 2x2 mean pool -> (B,64,4,48) f32
__global__ void avgpool_g(const float* __restrict__ g, float* __restrict__ o, int total) {
  int i = blockIdx.x * 256 + threadIdx.x; if (i >= total) return;
  int w = i % 48; int r = i / 48; int h = r % 4; r /= 4;
  int oc = r % 64; int b = r / 64;
  const float* base = g + ((size_t)b * 64 + oc) * 768;
  int n0 = (2 * h) * 96 + 2 * w;
  o[i] = 0.25f * (base[n0] + base[n0 + 1] + base[n0 + 96] + base[n0 + 97]);
}

// up1 = 1x1 conv (64 -> 5) on x1 (NHWC f16), out f32 NCHW
__global__ void up1_kernel(const h16* __restrict__ x1f, const float* __restrict__ w,
                           const float* __restrict__ bias, float* __restrict__ o,
                           int total) {
  int i = blockIdx.x * 256 + threadIdx.x; if (i >= total) return;
  const h16* f = x1f + (size_t)i * 64;
  int b = i / 768; int hw = i % 768;
  for (int oc = 0; oc < 5; ++oc) {
    float s = bias[oc];
    for (int c = 0; c < 64; ++c) s += (float)f[c] * w[oc * 64 + c];
    o[((size_t)b * 5 + oc) * 768 + hw] = s;
  }
}

// convt2 (64->5) + 1x1 conv (5->5) fused; h7 NHWC f16 (B,8,96,64) ->
// h9 f16 stored NCHW-flattened rows [(b*5+j)*16 + y][x] for the final GEMM.
__global__ void tail_kernel(const h16* __restrict__ h7, const float* __restrict__ w2,
                            const float* __restrict__ b2, const float* __restrict__ w3,
                            const float* __restrict__ b3, h16* __restrict__ h9,
                            int total) {
  int i = blockIdx.x * 256 + threadIdx.x; if (i >= total) return;
  int xx = i % 192; int r = i / 192; int y = r % 16; int b = r / 16;
  int h = y >> 1, p = y & 1, wq = xx >> 1, q = xx & 1;
  const h16* f = h7 + (((size_t)b * 8 + h) * 96 + wq) * 64;
  float h8v[5];
  for (int o = 0; o < 5; ++o) {
    float s = b2[o];
    for (int c = 0; c < 64; ++c)
      s += (float)f[c] * w2[(((size_t)c * 5 + o) * 2 + p) * 2 + q];
    h8v[o] = s;
  }
  for (int j = 0; j < 5; ++j) {
    float s = b3[j];
    #pragma unroll
    for (int k = 0; k < 5; ++k) s += h8v[k] * w3[j * 5 + k];
    h9[(((size_t)b * 5 + j) * 16 + y) * 192 + xx] = (h16)s;
  }
}

// ---------------------------------------------------------------------------
extern "C" void kernel_launch(void* const* d_in, const int* in_sizes, int n_in,
                              void* d_out, int out_size, void* d_ws, size_t ws_size,
                              hipStream_t stream) {
  (void)in_sizes; (void)n_in; (void)out_size; (void)ws_size;
  const float* x       = (const float*)d_in[0];
  const float* enc_w1  = (const float*)d_in[1];
  const float* enc_b1  = (const float*)d_in[2];
  const float* enc_w2  = (const float*)d_in[3];
  const float* enc_b2  = (const float*)d_in[4];
  const float* mid_w1  = (const float*)d_in[5];
  const float* mid_b1  = (const float*)d_in[6];
  const float* mid_w2  = (const float*)d_in[7];
  const float* mid_b2  = (const float*)d_in[8];
  const float* dec_w1  = (const float*)d_in[9];
  const float* dec_b1  = (const float*)d_in[10];
  const float* dec_w2  = (const float*)d_in[11];
  const float* dec_b2  = (const float*)d_in[12];
  const float* dect_w1 = (const float*)d_in[13];
  const float* dect_b1 = (const float*)d_in[14];
  const float* dect_w2 = (const float*)d_in[15];
  const float* dect_b2 = (const float*)d_in[16];
  const float* dec_w3  = (const float*)d_in[17];
  const float* dec_b3  = (const float*)d_in[18];
  const float* lin_w   = (const float*)d_in[19];
  const float* lin_b   = (const float*)d_in[20];
  const float* up1_w   = (const float*)d_in[21];
  const float* up1_b   = (const float*)d_in[22];
  const float* g1_w    = (const float*)d_in[23];
  const float* g1_b    = (const float*)d_in[24];
  const float* g2_w    = (const float*)d_in[25];
  const float* g2_b    = (const float*)d_in[26];
  float* out = (float*)d_out;

  // d_out layout (floats)
  const size_t UP1_OFF  = 1976320;
  const size_t X2_OFF   = 2467840;
  const size_t X1GP_OFF = 5613568;
  const size_t X2G_OFF  = 7186432;

  // ---- workspace bump allocator (~207 MB with aliasing) ----
  char* ws = (char*)d_ws; size_t off = 0;
  auto alloc = [&](size_t bytes) -> char* {
    char* p = ws + off; off = (off + bytes + 255) & ~(size_t)255; return p;
  };
  h16* xpad = (h16*)alloc((size_t)128*16*192*32*2);  // padded input
  h16* wp1  = (h16*)alloc((size_t)9*64*32*2);
  h16* wp2  = (h16*)alloc((size_t)9*64*64*2);
  h16* wpm1 = (h16*)alloc((size_t)9*128*64*2);
  h16* wpm2 = (h16*)alloc((size_t)9*128*128*2);
  h16* wpd1 = (h16*)alloc((size_t)9*64*128*2);
  h16* wpd2 = (h16*)alloc((size_t)9*64*64*2);
  h16* wt1p = (h16*)alloc((size_t)4*64*64*2);
  h16* wqr1 = (h16*)alloc((size_t)128*64*2);
  h16* wqr2 = (h16*)alloc((size_t)128*128*2);
  h16* lw16 = (h16*)alloc((size_t)193*192*2);
  h16* h1   = (h16*)alloc((size_t)50331648);   // (B,16,192,64)
  h16* h2   = (h16*)alloc((size_t)50331648);   // (B,16,192,64)
  h16* x1f  = (h16*)alloc((size_t)12582912);   // (B,8,96,64)
  h16* h3   = (h16*)alloc((size_t)25165824);   // (B,8,96,128)
  h16* h4   = (h16*)alloc((size_t)25165824);   // (B,8,96,128)
  h16* x2f  = (h16*)alloc((size_t)6291456);    // (B,4,48,128)
  h16* h5   = (h16*)alloc((size_t)3145728);    // (B,4,48,64)
  h16* h6   = (h16*)alloc((size_t)3145728);    // (B,4,48,64)
  h16* h9f  = (h16*)alloc((size_t)3932160);    // (B*5*16,192)
  // aliases (lifetimes do not overlap):
  float* qr1 = (float*)h1;                          // 98304 x 128 f32 (h1 dead)
  float* x1g = (float*)h2;                          // (B,64,768) f32 (h2 dead)
  float* qr2 = (float*)((char*)h2 + 25165824);      // 24576 x 128 f32
  float* sq1 = (float*)((char*)h2 + 25165824 + 12582912);
  float* sq2 = (float*)((char*)h2 + 25165824 + 12582912 + 393216);
  h16*   h7  = h4;                                  // (B,8,96,64) (h4 dead)

  int T;
  // ---- pack weights / input to f16 WMMA-ready layouts ----
  T = 128*16*192*32; pack_x<<<(T+255)/256,256,0,stream>>>(x, xpad, T);
  T = 9*64*32;   pack_conv3<<<(T+255)/256,256,0,stream>>>(enc_w1, wp1, 64, 5,   32,  T);
  T = 9*64*64;   pack_conv3<<<(T+255)/256,256,0,stream>>>(enc_w2, wp2, 64, 64,  64,  T);
  T = 9*128*64;  pack_conv3<<<(T+255)/256,256,0,stream>>>(mid_w1, wpm1,128, 64, 64,  T);
  T = 9*128*128; pack_conv3<<<(T+255)/256,256,0,stream>>>(mid_w2, wpm2,128, 128,128, T);
  T = 9*64*128;  pack_conv3<<<(T+255)/256,256,0,stream>>>(dec_w1, wpd1, 64, 128,128, T);
  T = 9*64*64;   pack_conv3<<<(T+255)/256,256,0,stream>>>(dec_w2, wpd2, 64, 64, 64,  T);
  T = 4*64*64;   pack_convt<<<(T+255)/256,256,0,stream>>>(dect_w1, wt1p, 64, 64, T);
  T = 128*64;    pack_qr<<<(T+255)/256,256,0,stream>>>(g1_w, wqr1, 64,  T);
  T = 128*128;   pack_qr<<<(T+255)/256,256,0,stream>>>(g2_w, wqr2, 128, T);
  T = 193*192;   cvt16<<<(T+255)/256,256,0,stream>>>(lin_w, lw16, T);

  // ---- encoder (M-tile 128, N-tile 64) ----
  conv3x3_wmma<true><<<dim3(3072,1),256,0,stream>>>(xpad, wp1, enc_b1, h1, 128,16,192,32,64);
  conv3x3_wmma<true><<<dim3(3072,1),256,0,stream>>>(h1,   wp2, enc_b2, h2, 128,16,192,64,64);
  T = 128*8*96*64;
  maxpool2h<<<(T+255)/256,256,0,stream>>>(h2, x1f, nullptr, 128,16,192,64, T);

  // ---- graph conv stage 1 (N=768, C=64) ----
  gemm_wmma<<<dim3(768,2),256,0,stream>>>(x1f, wqr1, nullptr, qr1, 98304, 128, 64);
  T = 98304; sqnorm<<<(T+255)/256,256,0,stream>>>(x1f, sq1, T, 64);
  graph_edge<<<dim3(12,128),64,0,stream>>>(x1f, sq1, qr1, g1_b, x1g, 768, 64);
  T = 128*64*4*48; avgpool_g<<<(T+255)/256,256,0,stream>>>(x1g, out + X1GP_OFF, T);

  // ---- mid ----
  conv3x3_wmma<true><<<dim3(768,2),256,0,stream>>>(x1f, wpm1, mid_b1, h3, 128,8,96,64,128);
  conv3x3_wmma<true><<<dim3(768,2),256,0,stream>>>(h3,  wpm2, mid_b2, h4, 128,8,96,128,128);
  T = 128*4*48*128;
  maxpool2h<<<(T+255)/256,256,0,stream>>>(h4, x2f, out + X2_OFF, 128,8,96,128, T);

  // ---- graph conv stage 2 (N=192, C=128) -> d_out directly ----
  gemm_wmma<<<dim3(192,2),256,0,stream>>>(x2f, wqr2, nullptr, qr2, 24576, 128, 128);
  T = 24576; sqnorm<<<(T+255)/256,256,0,stream>>>(x2f, sq2, T, 128);
  graph_edge<<<dim3(3,128),64,0,stream>>>(x2f, sq2, qr2, g2_b, out + X2G_OFF, 192, 128);

  // ---- decoder ----
  conv3x3_wmma<true><<<dim3(192,1),256,0,stream>>>(x2f, wpd1, dec_b1, h5, 128,4,48,128,64);
  conv3x3_wmma<true><<<dim3(192,1),256,0,stream>>>(h5,  wpd2, dec_b2, h6, 128,4,48,64,64);
  convt2x2_wmma<<<dim3(192,1,4),256,0,stream>>>(h6, wt1p, dect_b1, h7, 128,4,48,64,64);
  T = 128*16*192;
  tail_kernel<<<(T+255)/256,256,0,stream>>>(h7, dect_w2, dect_b2, dec_w3, dec_b3, h9f, T);

  // ---- final linear (M=10240, K=192, N=193) -> d_out[0:] ----
  gemm_wmma<<<dim3(80,4),256,0,stream>>>(h9f, lw16, lin_b, out, 10240, 193, 192);

  // ---- up1 1x1 conv -> d_out ----
  T = 128*8*96;
  up1_kernel<<<(T+255)/256,256,0,stream>>>(x1f, up1_w, up1_b, out + UP1_OFF, T);
}
